// Gdn3D_44908178047507
// MI455X (gfx1250) — matrix operand order repaired and checked by reading, not compiled, not run on hardware
//
#include <hip/hip_runtime.h>

typedef __attribute__((ext_vector_type(16))) _Float16 v16h;
typedef __attribute__((ext_vector_type(8)))  float    v8f;
typedef __attribute__((ext_vector_type(4)))  float    v4f;

#define NB   4
#define CH   64
#define DHW  (16 * 128 * 128)          // 262144 positions per (n, c)
#define TILE_P 128                     // positions per workgroup

// LDS staging in exact WMMA fragment layouts (lane-contiguous 32B chunks):
//   gA : ((kt*2+cb)*32 + lane)*16 + e  = kt*1024 + cb*512 + hi*256 + lr*16 + e
//   x2B: ((cb*8+pt)*32 + lane)*16 + e  = cb*4096 + pt*512 + hiB*256 + lr*16 + e
//   xD : ((kt*8+pt)*32 + lane)*8  + e  = kt*2048 + pt*256 + hiD*128 + lr*8 + e

__global__ __launch_bounds__(256) void gdn3d_wmma_kernel(
    const float* __restrict__ x,
    const float* __restrict__ gamma,
    const float* __restrict__ beta,
    float* __restrict__ out)
{
    __shared__ __align__(32) _Float16 gA[4096];    //  8 KB
    __shared__ __align__(32) _Float16 x2B[8192];   // 16 KB
    __shared__ __align__(32) float    xD[8192];    // 32 KB
    __shared__ float betas[CH];

    const int tid   = threadIdx.x;
    const int block = blockIdx.x;                 // 8192 blocks
    const int tiles_per_n = DHW / TILE_P;         // 2048
    const int n     = block / tiles_per_n;
    const int pbase = (block % tiles_per_n) * TILE_P;

    const float* xsrc = x   + (size_t)n * CH * DHW + pbase;
    float*       odst = out + (size_t)n * CH * DHW + pbase;

    // ---- Stage 0: gamma -> f16 A-fragment layout (imm-offset scatter) ----
    {
        const int k   = tid & 63;                 // fixed per thread
        const int c0g = tid >> 6;                 // 0..3
        const int kt  = k >> 4, lr = k & 15;
        const int gbase = kt * 1024 + lr * 16 + c0g;   // runtime base
        const float* gsrc = gamma + tid;               // + i*256 (imm)
        #pragma unroll
        for (int i = 0; i < 16; ++i) {
            const int cb = i >> 3;                     // compile-time
            const int hi = (i >> 1) & 1;               // compile-time
            const int eo = 4 * (i & 1) + 8 * ((i >> 2) & 1);
            gA[gbase + cb * 512 + hi * 256 + eo] = (_Float16)gsrc[i * 256];
        }
    }
    if (tid < CH) betas[tid] = beta[tid];

    // ---- Stage 1: coalesced b128 NT loads; imm-offset fragment scatter ----
    {
        const int c0 = tid >> 5;                  // 0..7
        const int p0 = (tid & 31) * 4;            // 0..124
        int x2base[4], xdbase[4];
        #pragma unroll
        for (int j = 0; j < 4; ++j) {
            const int p = p0 + j, pt = p >> 4, lr = p & 15;
            x2base[j] = pt * 512 + lr * 16 + c0;  // + cb*4096 + hiB*256 + 8*((i>>1)&1)
            xdbase[j] = pt * 256 + lr * 8  + c0;  // + kt*2048 + hiD*128
        }
        const float* src = xsrc + (size_t)c0 * DHW + p0;
        #pragma unroll
        for (int i = 0; i < 8; ++i) {
            const v4f v = __builtin_nontemporal_load(
                              (const v4f*)(src + (size_t)i * 8 * DHW));
            const int cb  = i >> 2;               // all compile-time
            const int hiB = i & 1;
            const int eo  = 8 * ((i >> 1) & 1);
            const int kt  = i >> 1;
            const int hiD = i & 1;
            #pragma unroll
            for (int j = 0; j < 4; ++j) {
                x2B[x2base[j] + cb * 4096 + hiB * 256 + eo] =
                    (_Float16)(v[j] * v[j]);
                xD[xdbase[j] + kt * 2048 + hiD * 128] = v[j];
            }
        }
    }
    __syncthreads();

    // ---- Stage 2+3: WMMA (beta folded into C), rsq, multiply, NT store ----
    const int wave = tid >> 5;                    // 0..7
    const int lane = tid & 31;
    const int lr   = lane & 15;
    const int hi   = lane >> 4;
    const int kt   = wave & 3;                    // k-tile (16 k each)
    const int ptg  = wave >> 2;                   // p-tile group (4 tiles each)

    const v16h a0 = *(const v16h*)&gA[(kt * 64 + lane) * 16];
    const v16h a1 = *(const v16h*)&gA[(kt * 64 + 32 + lane) * 16];

    v8f bv;                                       // beta in C/D layout (contiguous)
    #pragma unroll
    for (int e = 0; e < 8; ++e) bv[e] = betas[kt * 16 + e + (hi << 3)];

    float* obase = odst + (size_t)(kt * 16 + hi * 8) * DHW + ptg * 64 + lr;

    #pragma unroll
    for (int t = 0; t < 4; ++t) {
        const int pt = ptg * 4 + t;

        const v16h b0 = *(const v16h*)&x2B[       (pt * 32 + lane) * 16];
        const v16h b1 = *(const v16h*)&x2B[4096 + (pt * 32 + lane) * 16];

        v8f acc = bv;                             // D = gammaT*x2 + beta
        acc = __builtin_amdgcn_wmma_f32_16x16x32_f16(false, a0, false, b0,
                                                     (short)0, acc, false, false);
        acc = __builtin_amdgcn_wmma_f32_16x16x32_f16(false, a1, false, b1,
                                                     (short)0, acc, false, false);

        const v8f xv = *(const v8f*)&xD[kt * 2048 + pt * 256 + lane * 8];

        #pragma unroll
        for (int e = 0; e < 8; ++e) {
            const float r = __builtin_amdgcn_rsqf(acc[e]);   // raw v_rsq_f32
            __builtin_nontemporal_store(xv[e] * r,
                                        obase + (size_t)e * DHW + t * 16);
        }
    }
}

extern "C" void kernel_launch(void* const* d_in, const int* in_sizes, int n_in,
                              void* d_out, int out_size, void* d_ws, size_t ws_size,
                              hipStream_t stream) {
    const float* x     = (const float*)d_in[0];
    const float* gamma = (const float*)d_in[1];
    const float* beta  = (const float*)d_in[2];
    float* out = (float*)d_out;

    const int blocks = NB * (DHW / TILE_P);       // 8192
    gdn3d_wmma_kernel<<<blocks, 256, 0, stream>>>(x, gamma, beta, out);
}
